// DirectionalSelfAttention_66451734004218
// MI455X (gfx1250) — matrix operand order repaired and checked by reading, not compiled, not run on hardware
//
#include <hip/hip_runtime.h>
#include <hip/hip_bf16.h>
#include <math.h>

// ---------------------------------------------------------------------------
// Types for CDNA5 WMMA (wave32): V_WMMA_F32_16X16X32_BF16
// ---------------------------------------------------------------------------
typedef __bf16 bf16_t;
typedef __attribute__((ext_vector_type(16))) __bf16 v16bf;
typedef __attribute__((ext_vector_type(8)))  __bf16 v8bf;
typedef __attribute__((ext_vector_type(8)))  float  v8f;

#define WMMA_BF16(a, b, c) \
  __builtin_amdgcn_wmma_f32_16x16x32_bf16(false, (a), false, (b), (short)0, (c), false, false)

__device__ __forceinline__ bf16_t f2bf(float f) {
  unsigned u = __builtin_bit_cast(unsigned, f);
  unsigned r = (u + 0x7FFFu + ((u >> 16) & 1u)) >> 16;   // round-to-nearest-even
  unsigned short s = (unsigned short)r;
  return __builtin_bit_cast(bf16_t, s);
}

// A-fragment (16x32 bf16, row-major source, leading dim `ld`):
// lane L: row = L%16; K-chunks of 8 at (L/16)*8 and 16+(L/16)*8.
__device__ __forceinline__ v16bf load_a_frag(const bf16_t* base, int ld) {
  int lane = threadIdx.x & 31;
  int row  = lane & 15;
  int koff = (lane >> 4) * 8;
  union { v16bf v; v8bf h[2]; } u;
  u.h[0] = *(const v8bf*)(base + (size_t)row * ld + koff);        // 16B load
  u.h[1] = *(const v8bf*)(base + (size_t)row * ld + 16 + koff);   // 16B load
  return u.v;
}

// B-fragment (32x16 bf16) from K-major global memory (column n contiguous in K).
__device__ __forceinline__ v16bf load_b_frag(const bf16_t* base, int ld) {
  int lane = threadIdx.x & 31;
  int col  = lane & 15;
  int koff = (lane >> 4) * 8;
  const bf16_t* p = base + (size_t)col * ld + koff;
  union { v16bf v; v8bf h[2]; } u;
  u.h[0] = *(const v8bf*)(p);       // 16B
  u.h[1] = *(const v8bf*)(p + 16);  // 16B
  return u.v;
}

// B-fragment from an LDS tile staged column-major: 64 columns x 32 k, each
// column contiguous (32 bf16 = 64B).  colbase selects the 16-column group.
__device__ __forceinline__ v16bf lds_b_frag(const bf16_t* buf, int colbase) {
  int lane = threadIdx.x & 31;
  int col  = colbase + (lane & 15);
  int koff = (lane >> 4) * 8;
  const bf16_t* p = buf + col * 32 + koff;
  union { v16bf v; v8bf h[2]; } u;
  u.h[0] = *(const v8bf*)(p);       // ds_load_b128
  u.h[1] = *(const v8bf*)(p + 16);
  return u.v;
}

// Block-cooperative async copy of a 64x32 bf16 B tile (K-major source) into
// LDS via GLOBAL_LOAD_ASYNC_TO_LDS_B128 (tracked by ASYNCcnt).
// 4KB = 256 x 16B chunks; 128 threads x 2 chunks each.
__device__ __forceinline__ void async_b_tile_load(const bf16_t* gbase, int ldk,
                                                  bf16_t* lbuf) {
  int t = threadIdx.x;   // 0..127
#pragma unroll
  for (int i = 0; i < 2; ++i) {
    int c   = t + i * 128;
    int col = c >> 2;          // 0..63
    int q   = c & 3;           // 16B chunk within column
    unsigned long long ga =
        (unsigned long long)(uintptr_t)(gbase + (size_t)col * ldk + q * 8);
    unsigned la = (unsigned)(uintptr_t)(lbuf + col * 32 + q * 8);  // LDS offset
    asm volatile("global_load_async_to_lds_b128 %0, %1, off"
                 :: "v"(la), "v"(ga) : "memory");
  }
}

// ---------------------------------------------------------------------------
// fp32 -> bf16 conversion / weight transpose
// ---------------------------------------------------------------------------
__global__ void __launch_bounds__(256) cvt_kernel(const float* __restrict__ x,
                                                  bf16_t* __restrict__ y, int n) {
  int i = blockIdx.x * blockDim.x + threadIdx.x;
  if (i < n) y[i] = f2bf(x[i]);
}

// w[K][N] (row-major) -> wt[N][K] bf16
__global__ void __launch_bounds__(256) transpose_cvt_kernel(const float* __restrict__ w,
                                                            bf16_t* __restrict__ wt,
                                                            int K, int N) {
  int i = blockIdx.x * blockDim.x + threadIdx.x;
  if (i < K * N) {
    int k = i / N, n = i - k * N;
    wt[(size_t)n * K + k] = f2bf(w[i]);
  }
}

// ---------------------------------------------------------------------------
// QKV GEMM: [8192,1024] x [1024,3072] (+bias).  B tile (shared by the block's
// 4 waves) is double-buffered in LDS via async loads; A is software-pipelined
// in registers.  Epilogue scatters to Q[B,H,T,D], K[B,H,T,D], Vt[B,H,D,T].
// ---------------------------------------------------------------------------
__global__ void __launch_bounds__(128) gemm_qkv_kernel(
    const bf16_t* __restrict__ xb, const bf16_t* __restrict__ wT,
    const float* __restrict__ bias,
    bf16_t* __restrict__ Qb, bf16_t* __restrict__ Kb, bf16_t* __restrict__ Vt) {
  const int K = 1024;
  __shared__ alignas(16) bf16_t bstage[2][64 * 32];   // 2 x 4KB
  int lane = threadIdx.x & 31;
  int wave = threadIdx.x >> 5;
  int m0 = (blockIdx.x * 4 + wave) * 16;
  int n0 = blockIdx.y * 64;
  v8f acc0 = {}, acc1 = {}, acc2 = {}, acc3 = {};
  const bf16_t* abase = xb + (size_t)m0 * K;
  const bf16_t* bbase = wT + (size_t)n0 * K;

  async_b_tile_load(bbase, K, bstage[0]);             // prologue: buffer 0
  v16bf a_cur = load_a_frag(abase, K);
  int cur = 0;
  for (int kk = 0; kk < K; kk += 32) {
    asm volatile("s_wait_asynccnt 0x0" ::: "memory"); // my async writes landed
    __syncthreads();                                  // whole block's landed
    v16bf a_nxt = a_cur;
    if (kk + 32 < K) {
      async_b_tile_load(bbase + kk + 32, K, bstage[cur ^ 1]);  // overlap DMA
      a_nxt = load_a_frag(abase + kk + 32, K);                 // pipeline A
    }
    const bf16_t* bb = bstage[cur];
    // Batch all 8 ds_load_b128 before the WMMA group (partial dscnt waits).
    v16bf b0 = lds_b_frag(bb,  0);
    v16bf b1 = lds_b_frag(bb, 16);
    v16bf b2 = lds_b_frag(bb, 32);
    v16bf b3 = lds_b_frag(bb, 48);
    acc0 = WMMA_BF16(a_cur, b0, acc0);
    acc1 = WMMA_BF16(a_cur, b1, acc1);
    acc2 = WMMA_BF16(a_cur, b2, acc2);
    acc3 = WMMA_BF16(a_cur, b3, acc3);
    a_cur = a_nxt;
    cur ^= 1;
  }

  int hilo = lane >> 4, nlane = lane & 15;
#pragma unroll
  for (int r = 0; r < 8; ++r) {
    int mg = m0 + r + 8 * hilo;
    int b = mg >> 11, t = mg & 2047;
#pragma unroll
    for (int nt = 0; nt < 4; ++nt) {
      int c = n0 + nt * 16 + nlane;
      float v = (nt == 0 ? acc0[r] : nt == 1 ? acc1[r] : nt == 2 ? acc2[r] : acc3[r]) + bias[c];
      int which = c >> 10, cc = c & 1023, h = cc >> 6, d = cc & 63;
      size_t bh = (size_t)(b * 16 + h);
      if (which == 0)      Qb[(bh * 2048 + t) * 64 + d] = f2bf(v);
      else if (which == 1) Kb[(bh * 2048 + t) * 64 + d] = f2bf(v);
      else                 Vt[(bh * 64 + d) * 2048 + t] = f2bf(v);
    }
  }
}

// ---------------------------------------------------------------------------
// Flash attention: one wave per 16-query tile, 32-key tiles, online softmax.
// dir==1 -> anti-causal (k>=q), else causal (k<=q).  Output bf16 [B,T,C].
// ---------------------------------------------------------------------------
__global__ void __launch_bounds__(128) attn_kernel(
    const bf16_t* __restrict__ Qb, const bf16_t* __restrict__ Kb,
    const bf16_t* __restrict__ Vt, const int* __restrict__ dirp,
    bf16_t* __restrict__ Ob) {
  __shared__ alignas(16) bf16_t plds[4][16][40];   // per-wave P repack buffer
  int lane = threadIdx.x & 31;
  int wave = threadIdx.x >> 5;
  int id = blockIdx.x * 4 + wave;
  int qt = id & 127;          // 128 q-tiles per (b,h)
  int bh = id >> 7;           // 0..63
  int q0 = qt << 4;
  int dir = dirp[0];
  int hilo = lane >> 4, nlane = lane & 15;

  const bf16_t* qbase = Qb + ((size_t)bh * 2048 + q0) * 64;
  v16bf qf0 = load_a_frag(qbase, 64);        // d 0..31
  v16bf qf1 = load_a_frag(qbase + 32, 64);   // d 32..63

  v8f o0 = {}, o1 = {}, o2 = {}, o3 = {};
  float mrun[8], lrun[8];
#pragma unroll
  for (int r = 0; r < 8; ++r) { mrun[r] = -INFINITY; lrun[r] = 0.f; }

  int jstart, jend;
  if (dir == 1) { jstart = q0 >> 5; jend = 63; }
  else          { jstart = 0;       jend = (q0 + 15) >> 5; }

  for (int j = jstart; j <= jend; ++j) {
    const bf16_t* kbase = Kb + ((size_t)bh * 2048 + j * 32) * 64;
    v16bf kl0 = load_b_frag(kbase,                64);  // keys j*32..+15, d 0..31
    v16bf kl1 = load_b_frag(kbase + 32,           64);  // d 32..63
    v16bf kr0 = load_b_frag(kbase + 16 * 64,      64);  // keys +16..+31
    v16bf kr1 = load_b_frag(kbase + 16 * 64 + 32, 64);
    // Prefetch V fragments early so their loadcnt waits overlap the softmax.
    const bf16_t* vbase = Vt + (size_t)bh * 64 * 2048 + j * 32;
    v16bf vf0 = load_b_frag(vbase + (size_t)0 * 16 * 2048, 2048);
    v16bf vf1 = load_b_frag(vbase + (size_t)1 * 16 * 2048, 2048);
    v16bf vf2 = load_b_frag(vbase + (size_t)2 * 16 * 2048, 2048);
    v16bf vf3 = load_b_frag(vbase + (size_t)3 * 16 * 2048, 2048);

    v8f s0 = {}, s1 = {};
    s0 = WMMA_BF16(qf0, kl0, s0);
    s0 = WMMA_BF16(qf1, kl1, s0);
    s1 = WMMA_BF16(qf0, kr0, s1);
    s1 = WMMA_BF16(qf1, kr1, s1);

    bool needmask = (dir == 1) ? (j * 32 < q0 + 15) : (j * 32 + 31 > q0);
#pragma unroll
    for (int r = 0; r < 8; ++r) {
      float sv0 = s0[r] * 0.125f;        // 1/sqrt(64)
      float sv1 = s1[r] * 0.125f;
      if (needmask) {
        int qg = q0 + r + 8 * hilo;
        int kg = j * 32 + nlane;
        bool ok0 = (dir == 1) ? (kg >= qg) : (kg <= qg);
        bool ok1 = (dir == 1) ? (kg + 16 >= qg) : (kg + 16 <= qg);
        if (!ok0) sv0 = -INFINITY;
        if (!ok1) sv1 = -INFINITY;
      }
      // row max across 16 lanes (C-layout row lives in one 16-lane half)
      float mx = fmaxf(sv0, sv1);
#pragma unroll
      for (int off = 8; off >= 1; off >>= 1) mx = fmaxf(mx, __shfl_xor(mx, off, 16));
      float mnew  = fmaxf(mrun[r], mx);
      float alpha = exp2f((mrun[r] - mnew) * 1.44269504f);
      float p0 = exp2f((sv0 - mnew) * 1.44269504f);
      float p1 = exp2f((sv1 - mnew) * 1.44269504f);
      float rs = p0 + p1;
#pragma unroll
      for (int off = 8; off >= 1; off >>= 1) rs += __shfl_xor(rs, off, 16);
      lrun[r] = lrun[r] * alpha + rs;
      mrun[r] = mnew;
      o0[r] *= alpha; o1[r] *= alpha; o2[r] *= alpha; o3[r] *= alpha;
      // spill P (C-layout) to LDS for A-layout repack
      int row = r + 8 * hilo;
      plds[wave][row][nlane]      = f2bf(p0);
      plds[wave][row][16 + nlane] = f2bf(p1);
    }
    asm volatile("s_wait_dscnt 0x0" ::: "memory");  // LDS writes visible to our reads
    v16bf pf;
    {
      int prow = nlane;
      int koff = hilo * 8;
      union { v16bf v; v8bf h[2]; } u;
      u.h[0] = *(const v8bf*)&plds[wave][prow][koff];        // ds_load_b128
      u.h[1] = *(const v8bf*)&plds[wave][prow][16 + koff];
      pf = u.v;
    }
    // O += P(16x32) x V(32x64)
    o0 = WMMA_BF16(pf, vf0, o0);
    o1 = WMMA_BF16(pf, vf1, o1);
    o2 = WMMA_BF16(pf, vf2, o2);
    o3 = WMMA_BF16(pf, vf3, o3);
  }

  int b = bh >> 4, h = bh & 15;
#pragma unroll
  for (int r = 0; r < 8; ++r) {
    int t = q0 + r + 8 * hilo;
    float inv = 1.0f / lrun[r];
    size_t base = ((size_t)(b * 2048 + t)) * 1024 + (size_t)h * 64 + nlane;
    Ob[base +  0] = f2bf(o0[r] * inv);
    Ob[base + 16] = f2bf(o1[r] * inv);
    Ob[base + 32] = f2bf(o2[r] * inv);
    Ob[base + 48] = f2bf(o3[r] * inv);
  }
}

// ---------------------------------------------------------------------------
// Output projection: [8192,1024] x [1024,1024] + bias -> fp32 d_out.
// Same async-LDS double-buffered structure as the QKV GEMM.
// ---------------------------------------------------------------------------
__global__ void __launch_bounds__(128) gemm_proj_kernel(
    const bf16_t* __restrict__ ab, const bf16_t* __restrict__ wT,
    const float* __restrict__ bias, float* __restrict__ out) {
  const int K = 1024;
  __shared__ alignas(16) bf16_t bstage[2][64 * 32];
  int lane = threadIdx.x & 31;
  int wave = threadIdx.x >> 5;
  int m0 = (blockIdx.x * 4 + wave) * 16;
  int n0 = blockIdx.y * 64;
  v8f acc0 = {}, acc1 = {}, acc2 = {}, acc3 = {};
  const bf16_t* abase = ab + (size_t)m0 * K;
  const bf16_t* bbase = wT + (size_t)n0 * K;

  async_b_tile_load(bbase, K, bstage[0]);
  v16bf a_cur = load_a_frag(abase, K);
  int cur = 0;
  for (int kk = 0; kk < K; kk += 32) {
    asm volatile("s_wait_asynccnt 0x0" ::: "memory");
    __syncthreads();
    v16bf a_nxt = a_cur;
    if (kk + 32 < K) {
      async_b_tile_load(bbase + kk + 32, K, bstage[cur ^ 1]);
      a_nxt = load_a_frag(abase + kk + 32, K);
    }
    const bf16_t* bb = bstage[cur];
    v16bf b0 = lds_b_frag(bb,  0);
    v16bf b1 = lds_b_frag(bb, 16);
    v16bf b2 = lds_b_frag(bb, 32);
    v16bf b3 = lds_b_frag(bb, 48);
    acc0 = WMMA_BF16(a_cur, b0, acc0);
    acc1 = WMMA_BF16(a_cur, b1, acc1);
    acc2 = WMMA_BF16(a_cur, b2, acc2);
    acc3 = WMMA_BF16(a_cur, b3, acc3);
    a_cur = a_nxt;
    cur ^= 1;
  }

  int hilo = lane >> 4, nlane = lane & 15;
#pragma unroll
  for (int r = 0; r < 8; ++r) {
    int mg = m0 + r + 8 * hilo;
#pragma unroll
    for (int nt = 0; nt < 4; ++nt) {
      int c = n0 + nt * 16 + nlane;
      float v = (nt == 0 ? acc0[r] : nt == 1 ? acc1[r] : nt == 2 ? acc2[r] : acc3[r]) + bias[c];
      out[(size_t)mg * 1024 + c] = v;
    }
  }
}

// ---------------------------------------------------------------------------
// Host launcher
// ---------------------------------------------------------------------------
extern "C" void kernel_launch(void* const* d_in, const int* in_sizes, int n_in,
                              void* d_out, int out_size, void* d_ws, size_t ws_size,
                              hipStream_t stream) {
  (void)in_sizes; (void)n_in; (void)out_size; (void)ws_size;
  const float* x      = (const float*)d_in[0];
  const int*   dirp   = (const int*)d_in[1];
  const float* qkv_w  = (const float*)d_in[2];
  const float* qkv_b  = (const float*)d_in[3];
  const float* proj_w = (const float*)d_in[4];
  const float* proj_b = (const float*)d_in[5];
  float* out = (float*)d_out;

  char* ws = (char*)d_ws;
  size_t off = 0;
  auto alloc = [&](size_t bytes) -> void* {
    void* p = ws + off;
    off += (bytes + 255) & ~(size_t)255;
    return p;
  };
  bf16_t* xb     = (bf16_t*)alloc((size_t)8192 * 1024 * 2);  // x bf16 (reused as attn out)
  bf16_t* qkvwT  = (bf16_t*)alloc((size_t)3072 * 1024 * 2);  // qkv_w^T bf16
  bf16_t* projwT = (bf16_t*)alloc((size_t)1024 * 1024 * 2);  // proj_w^T bf16
  bf16_t* Qb     = (bf16_t*)alloc((size_t)8192 * 1024 * 2);  // [B,H,T,D]
  bf16_t* Kb     = (bf16_t*)alloc((size_t)8192 * 1024 * 2);  // [B,H,T,D]
  bf16_t* Vt     = (bf16_t*)alloc((size_t)8192 * 1024 * 2);  // [B,H,D,T]
  bf16_t* Ob     = xb;  // x bf16 is dead after the QKV GEMM

  cvt_kernel<<<(8192 * 1024) / 256, 256, 0, stream>>>(x, xb, 8192 * 1024);
  transpose_cvt_kernel<<<(1024 * 3072) / 256, 256, 0, stream>>>(qkv_w, qkvwT, 1024, 3072);
  transpose_cvt_kernel<<<(1024 * 1024) / 256, 256, 0, stream>>>(proj_w, projwT, 1024, 1024);
  gemm_qkv_kernel<<<dim3(128, 48), 128, 0, stream>>>(xb, qkvwT, qkv_b, Qb, Kb, Vt);
  attn_kernel<<<2048, 128, 0, stream>>>(Qb, Kb, Vt, dirp, Ob);
  gemm_proj_kernel<<<dim3(128, 16), 128, 0, stream>>>(Ob, projwT, proj_b, out);
}
